// CausalSelfAttention_56392920596738
// MI455X (gfx1250) — matrix operand order repaired and checked by reading, not compiled, not run on hardware
//
#include <hip/hip_runtime.h>
#include <stdint.h>

// Problem constants (match reference)
#define BB 4
#define SS 2048
#define DD 1024

typedef __attribute__((ext_vector_type(16))) __bf16 v16bf;
typedef __attribute__((ext_vector_type(8)))  float  v8f;

union Frag16 {
    v16bf   v;
    uint32_t w[8];
    uint4    q[2];
};

__device__ __forceinline__ unsigned short f32_to_bf16(float f) {
    unsigned int u = __float_as_uint(f);
    unsigned int r = 0x7FFFu + ((u >> 16) & 1u);  // round-to-nearest-even
    u += r;
    return (unsigned short)(u >> 16);
}

// A-matrix fragment (16x32, M x K), bf16, per ISA 7.12.2:
// lane L: row r = L&15; K-base c0 = (L>>4)*8; v0..v3 hold K=c0..c0+7, v4..v7 hold K=c0+16..c0+23.
// base points at tile element (m=0, k=0); row-major with leading dim ld (elements).
__device__ __forceinline__ v16bf load_frag_a(const unsigned short* base, int ld, int lane) {
    const int r  = lane & 15;
    const int c0 = (lane >> 4) * 8;
    const unsigned short* p = base + (size_t)r * ld + c0;
    Frag16 f;
    f.q[0] = *(const uint4*)(p);
    f.q[1] = *(const uint4*)(p + 16);
    return f.v;
}

// B-matrix fragment (32x16, K x N), bf16:
// lane L: col n = L&15; K-base k0 = (L>>4)*8; v0..v3 hold K=k0..k0+7, v4..v7 hold K=k0+16..k0+23 (2 per VGPR, low half first).
// Memory is addressed N-major: element (k,n) at base + n*ld + k (two runs of 8 contiguous k per lane).
__device__ __forceinline__ v16bf load_frag_b(const unsigned short* base, int ld, int lane) {
    const int n  = lane & 15;
    const int k0 = (lane >> 4) * 8;
    const unsigned short* p = base + (size_t)n * ld + k0;
    Frag16 f;
    f.q[0] = *(const uint4*)(p);
    f.q[1] = *(const uint4*)(p + 16);
    return f.v;
}

__device__ __forceinline__ v8f wmma_bf16(v16bf a, v16bf b, v8f c) {
    return __builtin_amdgcn_wmma_f32_16x16x32_bf16(false, a, false, b, (short)0, c, false, false);
}

// ---------------------------------------------------------------------------
// f32 -> bf16 elementwise convert
// ---------------------------------------------------------------------------
__global__ void cvt_bf16_kernel(const float* __restrict__ in,
                                unsigned short* __restrict__ out, int n) {
    int i = blockIdx.x * blockDim.x + threadIdx.x;
    int stride = gridDim.x * blockDim.x;
    for (; i < n; i += stride) out[i] = f32_to_bf16(in[i]);
}

// ---------------------------------------------------------------------------
// Projection GEMM: Out[row, e] = sum_d X[row, d] * W[e, d]   (row = b*S + s)
// X: [B*S, D] bf16 row-major. W: [D, D] bf16 row-major (e-major) == ideal B operand.
// Each wave computes a 64x64 tile. transpose_v: store Vt[b][e][s] instead.
// Grid: (B*S/64, D/256), block 128 (4 waves, wave w -> col offset w*64).
// ---------------------------------------------------------------------------
__global__ __launch_bounds__(128) void proj_kernel(
        const unsigned short* __restrict__ X,
        const unsigned short* __restrict__ W,
        unsigned short* __restrict__ Out,
        int transpose_v) {
    const int lane = threadIdx.x & 31;
    const int wave = threadIdx.x >> 5;
    const int r0 = blockIdx.x * 64;
    const int c0 = blockIdx.y * 256 + wave * 64;

    v8f zero;
    #pragma unroll
    for (int i = 0; i < 8; ++i) zero[i] = 0.0f;
    v8f acc[4][4];
    #pragma unroll
    for (int i = 0; i < 4; ++i)
        #pragma unroll
        for (int j = 0; j < 4; ++j) acc[i][j] = zero;

    for (int k = 0; k < DD; k += 32) {
        v16bf a[4], bf[4];
        #pragma unroll
        for (int i = 0; i < 4; ++i)
            a[i] = load_frag_a(X + (size_t)(r0 + i * 16) * DD + k, DD, lane);
        #pragma unroll
        for (int j = 0; j < 4; ++j)
            bf[j] = load_frag_b(W + (size_t)(c0 + j * 16) * DD + k, DD, lane);
        #pragma unroll
        for (int i = 0; i < 4; ++i)
            #pragma unroll
            for (int j = 0; j < 4; ++j)
                acc[i][j] = wmma_bf16(a[i], bf[j], acc[i][j]);
    }

    // C/D layout: lane holds col n = lane&15; VGPR t holds row mb + t, mb = (lane>>4)*8.
    const int n  = lane & 15;
    const int mb = (lane >> 4) * 8;
    #pragma unroll
    for (int i = 0; i < 4; ++i) {
        #pragma unroll
        for (int j = 0; j < 4; ++j) {
            #pragma unroll
            for (int t = 0; t < 8; ++t) {
                const int row = r0 + i * 16 + mb + t;   // global b*S + s
                const int e   = c0 + j * 16 + n;
                const unsigned short h = f32_to_bf16(acc[i][j][t]);
                if (!transpose_v) {
                    Out[(size_t)row * DD + e] = h;
                } else {
                    const int b = row >> 11;            // / S
                    const int s = row & (SS - 1);       // % S
                    Out[((size_t)b * DD + e) * SS + s] = h;  // Vt[b][e][s]
                }
            }
        }
    }
}

// ---------------------------------------------------------------------------
// Flash attention: one workgroup (4 waves) per (batch, 16-query tile).
// Waves split D: D-slice of 256 each, both for score K-reduction (partials
// reduced in LDS) and for the PV output accumulator (16x256 f32 = 128 VGPRs).
// Online softmax state (m, l, alpha) per query row in LDS.
// ---------------------------------------------------------------------------
__global__ __launch_bounds__(128) void attn_kernel(
        const unsigned short* __restrict__ Qb,   // [B][S][D] bf16
        const unsigned short* __restrict__ Kb,   // [B][S][D] bf16
        const unsigned short* __restrict__ Vt,   // [B][D][S] bf16 (transposed)
        float* __restrict__ Out) {               // [B][S][D] f32
    __shared__ float sPart[4 * 16 * 32];   // per-wave partial score tiles
    __shared__ float sSc[16 * 32];         // reduced scores -> exp'ed P (f32)
    __shared__ float sM[16], sL[16], sA[16];

    const int lane = threadIdx.x & 31;
    const int wave = threadIdx.x >> 5;
    const int tile = blockIdx.x;           // 0 .. B*(S/16)-1
    const int b    = tile >> 7;            // S/16 = 128 tiles per batch
    const int q0   = (tile & 127) * 16;
    const size_t qkBase = (size_t)b * SS * DD;
    const size_t vBase  = (size_t)b * DD * SS;
    const int ds = wave * 256;             // this wave's D-slice

    if (threadIdx.x < 16) { sM[threadIdx.x] = -__builtin_inff(); sL[threadIdx.x] = 0.0f; }
    __syncthreads();

    // Preload Q A-frags for this wave's D-slice (reused every key chunk).
    v16bf qf[8];
    #pragma unroll
    for (int kk = 0; kk < 8; ++kk)
        qf[kk] = load_frag_a(Qb + qkBase + (size_t)q0 * DD + ds + kk * 32, DD, lane);

    v8f zero;
    #pragma unroll
    for (int i = 0; i < 8; ++i) zero[i] = 0.0f;
    v8f acc[16];
    #pragma unroll
    for (int i = 0; i < 16; ++i) acc[i] = zero;

    const int n  = lane & 15;
    const int mb = (lane >> 4) * 8;

    const int nchunk = (q0 + 15) / 32 + 1;  // causal: only chunks with j0 <= q0+15
    for (int ch = 0; ch < nchunk; ++ch) {
        const int j0 = ch * 32;

        // ---- 1: partial scores over own D-slice: S_part = Q_slice * K_slice^T
        #pragma unroll
        for (int nt = 0; nt < 2; ++nt) {
            v8f sa = zero;
            #pragma unroll
            for (int kk = 0; kk < 8; ++kk) {
                v16bf kf = load_frag_b(
                    Kb + qkBase + (size_t)(j0 + nt * 16) * DD + ds + kk * 32, DD, lane);
                sa = wmma_bf16(qf[kk], kf, sa);
            }
            #pragma unroll
            for (int t = 0; t < 8; ++t)
                sPart[wave * 512 + (mb + t) * 32 + nt * 16 + n] = sa[t];
        }
        __syncthreads();

        // ---- 2: reduce partials across waves, scale 1/sqrt(D), causal mask
        {
            const int e0 = threadIdx.x * 4;
            #pragma unroll
            for (int t = 0; t < 4; ++t) {
                const int e = e0 + t;
                const int row = e >> 5, col = e & 31;
                float v = sPart[e] + sPart[512 + e] + sPart[1024 + e] + sPart[1536 + e];
                v *= 0.03125f;  // 1/sqrt(1024)
                if (j0 + col > q0 + row) v = -__builtin_inff();
                sSc[e] = v;
            }
        }
        __syncthreads();

        // ---- 3: online softmax stats (one lane per query row)
        if (threadIdx.x < 16) {
            const int r = threadIdx.x;
            const float mOld = sM[r];
            float cm = -__builtin_inff();
            #pragma unroll 4
            for (int j = 0; j < 32; ++j) cm = fmaxf(cm, sSc[r * 32 + j]);
            const float mNew  = fmaxf(mOld, cm);
            const float alpha = __expf(mOld - mNew);
            float sum = 0.0f;
            #pragma unroll 4
            for (int j = 0; j < 32; ++j) {
                const float p = __expf(sSc[r * 32 + j] - mNew);
                sSc[r * 32 + j] = p;
                sum += p;
            }
            sL[r] = sL[r] * alpha + sum;
            sM[r] = mNew;
            sA[r] = alpha;
        }
        __syncthreads();

        // ---- 4: rescale accumulator, build P A-frag, PV GEMM over D-slice
        {
            float al[8];
            #pragma unroll
            for (int t = 0; t < 8; ++t) al[t] = sA[mb + t];
            #pragma unroll
            for (int t2 = 0; t2 < 16; ++t2)
                #pragma unroll
                for (int t = 0; t < 8; ++t) acc[t2][t] *= al[t];

            // P fragment (16x32 A-matrix layout) from exp'ed scores in LDS
            Frag16 pf;
            const int r  = lane & 15;
            const int c0 = (lane >> 4) * 8;
            #pragma unroll
            for (int i = 0; i < 4; ++i) {
                const unsigned short lo  = f32_to_bf16(sSc[r * 32 + c0 + 2 * i]);
                const unsigned short hi  = f32_to_bf16(sSc[r * 32 + c0 + 2 * i + 1]);
                pf.w[i] = (uint32_t)lo | ((uint32_t)hi << 16);
                const unsigned short lo2 = f32_to_bf16(sSc[r * 32 + c0 + 16 + 2 * i]);
                const unsigned short hi2 = f32_to_bf16(sSc[r * 32 + c0 + 16 + 2 * i + 1]);
                pf.w[4 + i] = (uint32_t)lo2 | ((uint32_t)hi2 << 16);
            }

            #pragma unroll
            for (int t2 = 0; t2 < 16; ++t2) {
                // B-frag from Vt: element (k=key j0+k, n=e) at Vt[b][e][j0+k] (contig in k)
                v16bf vf = load_frag_b(
                    Vt + vBase + (size_t)(ds + t2 * 16) * SS + j0, SS, lane);
                acc[t2] = wmma_bf16(pf.v, vf, acc[t2]);
            }
        }
    }

    // ---- epilogue: normalize by 1/l and store f32 output
    {
        float invl[8];
        #pragma unroll
        for (int t = 0; t < 8; ++t) invl[t] = 1.0f / sL[mb + t];
        #pragma unroll
        for (int t2 = 0; t2 < 16; ++t2) {
            #pragma unroll
            for (int t = 0; t < 8; ++t) {
                const size_t row = (size_t)b * SS + q0 + mb + t;
                Out[row * DD + ds + t2 * 16 + n] = acc[t2][t] * invl[t];
            }
        }
    }
}

// ---------------------------------------------------------------------------
extern "C" void kernel_launch(void* const* d_in, const int* in_sizes, int n_in,
                              void* d_out, int out_size, void* d_ws, size_t ws_size,
                              hipStream_t stream) {
    const float* x  = (const float*)d_in[0];
    const float* Wq = (const float*)d_in[1];
    const float* Wk = (const float*)d_in[2];
    const float* Wv = (const float*)d_in[3];
    float* out = (float*)d_out;
    char* ws = (char*)d_ws;

    const size_t NX = (size_t)BB * SS * DD;     // 8,388,608
    const size_t NW = (size_t)DD * DD;          // 1,048,576

    unsigned short* xb  = (unsigned short*)(ws);
    unsigned short* wqb = (unsigned short*)(ws + NX * 2);
    unsigned short* wkb = (unsigned short*)(ws + NX * 2 + NW * 2);
    unsigned short* wvb = (unsigned short*)(ws + NX * 2 + NW * 4);
    unsigned short* Qb  = (unsigned short*)(ws + NX * 2 + NW * 6);
    unsigned short* Kb  = (unsigned short*)(ws + NX * 4 + NW * 6);
    unsigned short* Vt  = (unsigned short*)(ws + NX * 6 + NW * 6);
    // total workspace: 4*NX*2 + 3*NW*2 = 73,400,320 bytes

    cvt_bf16_kernel<<<4096, 256, 0, stream>>>(x,  xb,  (int)NX);
    cvt_bf16_kernel<<<1024, 256, 0, stream>>>(Wq, wqb, (int)NW);
    cvt_bf16_kernel<<<1024, 256, 0, stream>>>(Wk, wkb, (int)NW);
    cvt_bf16_kernel<<<1024, 256, 0, stream>>>(Wv, wvb, (int)NW);

    dim3 pg(BB * SS / 64, DD / 256);  // (128, 4)
    proj_kernel<<<pg, 128, 0, stream>>>(xb, wqb, Qb, 0);
    proj_kernel<<<pg, 128, 0, stream>>>(xb, wkb, Kb, 0);
    proj_kernel<<<pg, 128, 0, stream>>>(xb, wvb, Vt, 1);

    attn_kernel<<<BB * (SS / 16), 128, 0, stream>>>(Qb, Kb, Vt, out);
}
